// Megnet_NodeModel_74818330296971
// MI455X (gfx1250) — matrix operand order repaired and checked by reading, not compiled, not run on hardware
//
#include <hip/hip_runtime.h>

typedef __attribute__((ext_vector_type(16))) _Float16 v16h;
typedef __attribute__((ext_vector_type(8)))  _Float16 v8h;
typedef __attribute__((ext_vector_type(8)))  float    v8f;

namespace {
constexpr int    kDim      = 128;
constexpr int    kNodes    = 50000;
constexpr int    kEdges    = 600000;
constexpr float  kEps      = 1e-5f;
constexpr int    kRowTiles = kNodes / 16;          // 3125 (exact)

// float region of workspace (offsets in floats)
constexpr size_t OF_SUMS  = 0;                               // 50000*128 edge-sums, reused as h
constexpr size_t OF_CNT   = OF_SUMS + (size_t)kNodes * kDim; // 6,400,000
constexpr size_t OF_STATS = OF_CNT + (size_t)kNodes;         // 6,450,000 ; 3 layers * 512
constexpr size_t F_TOTAL  = OF_STATS + 3 * 512;              // 6,451,536 floats

// half region starts at byte HB (32B aligned: 25,806,144)
constexpr size_t HB      = F_TOTAL * 4;
constexpr size_t OH_COMB = 0;                                 // 50000*384 halfs
constexpr size_t OH_HF   = OH_COMB + (size_t)kNodes * 384;    // 50000*128 halfs
constexpr size_t OH_W0   = OH_HF + (size_t)kNodes * kDim;
constexpr size_t OH_W1   = OH_W0 + 128 * 384;
constexpr size_t OH_W2   = OH_W1 + 128 * 128;
} // namespace

__global__ void zero_f32(float* __restrict__ p, size_t n) {
  size_t i = (size_t)blockIdx.x * blockDim.x + threadIdx.x;
  size_t stride = (size_t)gridDim.x * blockDim.x;
  for (; i < n; i += stride) p[i] = 0.0f;
}

// 128 threads per edge; atomic scatter-add of edge features + counts.
__global__ void edge_scatter(const float* __restrict__ edge_attr,
                             const long long* __restrict__ edge_index,
                             float* __restrict__ sums, float* __restrict__ cnt) {
  int e = blockIdx.x * 2 + (threadIdx.x >> 7);
  if (e >= kEdges) return;
  int t = threadIdx.x & 127;
  long long s = edge_index[e];                     // edge_index[0][e] (src row)
  atomicAdd(&sums[(size_t)s * kDim + t], edge_attr[(size_t)e * kDim + t]);
  if (t == 0) atomicAdd(&cnt[s], 1.0f);
}

// comb[n] = [x[n] | sums[n]/max(cnt,1) | u[batch[n]]] in f16
__global__ void prep_comb(const float* __restrict__ x, const float* __restrict__ sums,
                          const float* __restrict__ cnt, const float* __restrict__ u,
                          const long long* __restrict__ batch,
                          _Float16* __restrict__ comb) {
  int n = blockIdx.x * 2 + (threadIdx.x >> 7);
  if (n >= kNodes) return;
  int t = threadIdx.x & 127;
  _Float16* row = comb + (size_t)n * 384;
  row[t] = (_Float16)x[(size_t)n * kDim + t];
  float inv = 1.0f / fmaxf(cnt[n], 1.0f);
  row[128 + t] = (_Float16)(sums[(size_t)n * kDim + t] * inv);
  long long g = batch[n];
  row[256 + t] = (_Float16)u[(size_t)g * kDim + t];
}

__global__ void conv_weights(const float* __restrict__ W0, const float* __restrict__ W1,
                             const float* __restrict__ W2, _Float16* __restrict__ w) {
  int i = blockIdx.x * blockDim.x + threadIdx.x;
  if (i >= 81920) return;
  float v;
  if (i < 49152)      v = W0[i];
  else if (i < 65536) v = W1[i - 49152];
  else                v = W2[i - 65536];
  w[i] = (_Float16)v;
}

// One wave computes a full 16x128 output tile: out = relu(A @ W^T + b),
// with fused per-channel sum / sum-of-squares accumulation for BatchNorm.
// A: [N, K] f16 row-major ; W: [128, K] f16 row-major (B[k][j] = W[j][k]).
template <int K>
__global__ void __launch_bounds__(256)
gemm_relu_stats(const _Float16* __restrict__ A, const _Float16* __restrict__ W,
                const float* __restrict__ bias, float* __restrict__ out,
                float* __restrict__ ssum, float* __restrict__ ssq) {
  int wid = (int)((blockIdx.x * blockDim.x + threadIdx.x) >> 5);
  if (wid >= kRowTiles) return;                    // whole-wave exit: EXEC all-1s for WMMA
  int lane = threadIdx.x & 31;
  int col  = lane & 15;                            // N index within 16-col tile
  int hg   = lane >> 4;                            // half-wave group

  const _Float16* Arow = A + (size_t)(wid * 16 + col) * K;

  v8f acc[8];
#pragma unroll
  for (int j = 0; j < 8; ++j) {
    float bv = bias[j * 16 + col];                 // bias depends on channel only
#pragma unroll
    for (int r = 0; r < 8; ++r) acc[j][r] = bv;
  }

  union AU { v16h v; v8h h[2]; };
#pragma unroll
  for (int kt = 0; kt < K / 32; ++kt) {
    // 16-bit A operand layout: halves 0-15 <- K=kb+hg*16.. (contig, matches B)
    AU a;
    a.v = *(const v16h*)(Arow + kt * 32 + hg * 16);
#pragma unroll
    for (int j = 0; j < 8; ++j) {
      // B operand: lane holds B[kb+hg*16 .. +15][col] = W[col][kb+hg*16 .. +15] (contig)
      v16h b = *(const v16h*)(W + (size_t)(j * 16 + col) * K + kt * 32 + hg * 16);
      acc[j] = __builtin_amdgcn_wmma_f32_16x16x32_f16(
          false, a.v, false, b, (short)0, acc[j], false, false);
    }
  }

  int row0 = wid * 16;
#pragma unroll
  for (int j = 0; j < 8; ++j) {
    float s = 0.f, s2 = 0.f;
#pragma unroll
    for (int r = 0; r < 8; ++r) {                  // C layout: M = r + hg*8, N = col
      float v = acc[j][r];
      v = v > 0.f ? v : 0.f;                       // ReLU
      out[(size_t)(row0 + hg * 8 + r) * kDim + j * 16 + col] = v;
      s += v; s2 += v * v;
    }
    s  += __shfl_xor(s, 16, 32);                   // combine the two half-wave groups
    s2 += __shfl_xor(s2, 16, 32);
    if (hg == 0) {
      atomicAdd(&ssum[j * 16 + col], s);
      atomicAdd(&ssq[j * 16 + col], s2);
    }
  }
}

// stats layout per layer: [sum(128) | sumsq(128) | scale(128) | shift(128)]
__global__ void bn_finalize(float* __restrict__ stats, const float* __restrict__ gamma,
                            const float* __restrict__ beta) {
  int c = threadIdx.x;                             // 128 threads
  float mean = stats[c] * (1.0f / kNodes);
  float var  = stats[128 + c] * (1.0f / kNodes) - mean * mean;
  float rs   = rsqrtf(var + kEps);
  float sc   = gamma[c] * rs;
  stats[256 + c] = sc;
  stats[384 + c] = beta[c] - sc * mean;
}

__global__ void bn_apply_f16(const float* __restrict__ h, const float* __restrict__ stats,
                             _Float16* __restrict__ o) {
  size_t n = (size_t)kNodes * kDim;
  size_t stride = (size_t)gridDim.x * blockDim.x;
  for (size_t i = (size_t)blockIdx.x * blockDim.x + threadIdx.x; i < n; i += stride) {
    int c = (int)(i & 127);
    o[i] = (_Float16)(h[i] * stats[256 + c] + stats[384 + c]);
  }
}

__global__ void bn_apply_f32(const float* __restrict__ h, const float* __restrict__ stats,
                             float* __restrict__ o) {
  size_t n = (size_t)kNodes * kDim;
  size_t stride = (size_t)gridDim.x * blockDim.x;
  for (size_t i = (size_t)blockIdx.x * blockDim.x + threadIdx.x; i < n; i += stride) {
    int c = (int)(i & 127);
    o[i] = h[i] * stats[256 + c] + stats[384 + c];
  }
}

extern "C" void kernel_launch(void* const* d_in, const int* in_sizes, int n_in,
                              void* d_out, int out_size, void* d_ws, size_t ws_size,
                              hipStream_t stream) {
  (void)in_sizes; (void)n_in; (void)out_size; (void)ws_size;
  const float*     x          = (const float*)d_in[0];
  const long long* edge_index = (const long long*)d_in[1];
  const float*     edge_attr  = (const float*)d_in[2];
  const float*     u          = (const float*)d_in[3];
  const long long* batch      = (const long long*)d_in[4];
  const float*     W0 = (const float*)d_in[5];
  const float*     b0 = (const float*)d_in[6];
  const float*     W1 = (const float*)d_in[7];
  const float*     b1 = (const float*)d_in[8];
  const float*     W2 = (const float*)d_in[9];
  const float*     b2 = (const float*)d_in[10];
  const float*     g0 = (const float*)d_in[11];
  const float*     be0 = (const float*)d_in[12];
  const float*     g1 = (const float*)d_in[13];
  const float*     be1 = (const float*)d_in[14];
  const float*     g2 = (const float*)d_in[15];
  const float*     be2 = (const float*)d_in[16];

  float* fws   = (float*)d_ws;
  float* sums  = fws + OF_SUMS;     // edge sums; reused as h0/h1/h2
  float* cnt   = fws + OF_CNT;
  float* stats = fws + OF_STATS;
  _Float16* hws  = (_Float16*)((char*)d_ws + HB);
  _Float16* comb = hws + OH_COMB;
  _Float16* hf   = hws + OH_HF;
  _Float16* w0h  = hws + OH_W0;
  _Float16* w1h  = hws + OH_W1;
  _Float16* w2h  = hws + OH_W2;
  float* h = sums;                  // alias: scatter buffer reused as activations

  zero_f32<<<2048, 256, 0, stream>>>(fws, F_TOTAL);
  edge_scatter<<<(kEdges + 1) / 2, 256, 0, stream>>>(edge_attr, edge_index, sums, cnt);
  prep_comb<<<(kNodes + 1) / 2, 256, 0, stream>>>(x, sums, cnt, u, batch, comb);
  conv_weights<<<(81920 + 255) / 256, 256, 0, stream>>>(W0, W1, W2, w0h);

  int gblocks = (kRowTiles + 7) / 8;               // 8 waves (row tiles) per 256-thread block
  gemm_relu_stats<384><<<gblocks, 256, 0, stream>>>(comb, w0h, b0, h, stats + 0, stats + 128);
  bn_finalize<<<1, 128, 0, stream>>>(stats, g0, be0);
  bn_apply_f16<<<1024, 256, 0, stream>>>(h, stats, hf);

  gemm_relu_stats<128><<<gblocks, 256, 0, stream>>>(hf, w1h, b1, h, stats + 512, stats + 640);
  bn_finalize<<<1, 128, 0, stream>>>(stats + 512, g1, be1);
  bn_apply_f16<<<1024, 256, 0, stream>>>(h, stats + 512, hf);

  gemm_relu_stats<128><<<gblocks, 256, 0, stream>>>(hf, w2h, b2, h, stats + 1024, stats + 1152);
  bn_finalize<<<1, 128, 0, stream>>>(stats + 1024, g2, be2);
  bn_apply_f32<<<1024, 256, 0, stream>>>(h, stats + 1024, (float*)d_out);
}
